// SNNlifadex_7516192768084
// MI455X (gfx1250) — matrix operand order repaired and checked by reading, not compiled, not run on hardware
//
#include <hip/hip_runtime.h>
#include <cstdint>

typedef __bf16 bf16_t;
typedef bf16_t v16bf __attribute__((ext_vector_type(16)));
typedef bf16_t v8bf  __attribute__((ext_vector_type(8)));
typedef float  v8f   __attribute__((ext_vector_type(8)));
typedef float  v4f   __attribute__((ext_vector_type(4)));

#define DTs            0.001f
#define TAU_FILT_INV   223.1435511314f
#define TAU_SYN_INV    200.0f
#define TAU_MEM_INV    100.0f
#define TAU_ADA_INV    2.0f
#define ADAPT_CURRENT  4.0f
#define ADAPT_SPIKE    0.02f
#define DELTA_T        0.5f
#define V_TH           1.0f
#define V_LEAK         0.0f
#define V_RESET        0.0f

#define T_STEPS 500
#define BATCH   64
#define FDIM    256
#define HDIM    512
#define ODIM    128
#define ROWS    (T_STEPS * BATCH)   // 32000

// ---------------- workspace layout (bytes) ----------------
static const size_t OFF_WINP  = 0;                               // 256*512*2
static const size_t OFF_WRECP = OFF_WINP  + (size_t)FDIM*HDIM*2; // 512*512*2
static const size_t OFF_WOUTP = OFF_WRECP + (size_t)HDIM*HDIM*2; // 512*128*2
static const size_t OFF_XB    = OFF_WOUTP + (size_t)HDIM*ODIM*2; // bf16 [32000][256]
static const size_t OFF_XPROJ = OFF_XB    + (size_t)ROWS*FDIM*2; // f32 [32000][512]
static const size_t OFF_SPK   = OFF_XPROJ + (size_t)ROWS*HDIM*4; // bf16 [32000][512]
static const size_t OFF_LIN   = OFF_SPK   + (size_t)ROWS*HDIM*2; // f32 [32000][128]
static const size_t OFF_SYNC  = OFF_LIN   + (size_t)ROWS*ODIM*4; // 2 x u32

// ---------------- fragment loaders ----------------
// A fragment (16x32 bf16): lane l (<16): row m=l, K halves kb..kb+7 and kb+16..kb+23,
// kb = k0 (+8 for lanes >= 16).
__device__ __forceinline__ v16bf load_a_bf16(const bf16_t* base, int ld, int mbase,
                                             int k0, int lane) {
  int row = mbase + (lane & 15);
  int kb  = k0 + ((lane >> 4) << 3);
  const bf16_t* p = base + (size_t)row * ld + kb;
  v8bf lo = *(const v8bf*)p;
  v8bf hi = *(const v8bf*)(p + 16);
  v16bf r;
  #pragma unroll
  for (int i = 0; i < 8; ++i) { r[i] = lo[i]; r[i + 8] = hi[i]; }
  return r;
}

// B fragment from fragment-major packed array: frag (tk,tn), lane reads 16
// contiguous bf16 (32B) -> two global_load_b128.
__device__ __forceinline__ v16bf load_b_pk(const bf16_t* pk, int NT, int tk, int tn,
                                           int lane) {
  return *(const v16bf*)(pk + (((size_t)tk * NT + tn) * 32 + lane) * 16);
}

__device__ __forceinline__ v8f wmma_bf16(v16bf a, v16bf b, v8f c) {
  return __builtin_amdgcn_wmma_f32_16x16x32_bf16(false, a, false, b, (short)0, c,
                                                 false, false);
}

// ---------------- kernel 0a: repack W[N][K] f32 -> fragment-major bf16 ----------------
// packed element (tk,tn,lane,i) = W[n][k], n = tn*16 + (lane&15),
// k = tk*32 + ((lane>>4)<<3) + (i&7) + ((i>>3)<<4)   (B 32x16 layout: lanes<16 K=0..7,16..23, lanes>=16 K=8..15,24..31)
__global__ void pack_b_kernel(const float* __restrict__ w, bf16_t* __restrict__ pk,
                              int K, int N) {
  int gid = blockIdx.x * 256 + threadIdx.x;
  if (gid >= K * N) return;
  int i    = gid & 15;
  int lane = (gid >> 4) & 31;
  int fi   = gid >> 9;
  int NT   = N >> 4;
  int tn   = fi % NT;
  int tk   = fi / NT;
  int n = tn * 16 + (lane & 15);
  int k = tk * 32 + ((lane >> 4) << 3) + (i & 7) + ((i >> 3) << 4);
  pk[gid] = (bf16_t)w[(size_t)n * K + k];
}

// ---------------- kernel 0b: x f32 -> bf16 (row-major, converted once) ----------------
__global__ void cvt_x_kernel(const float* __restrict__ src, bf16_t* __restrict__ dst) {
  int gid = blockIdx.x * 256 + threadIdx.x;       // one float4 per thread
  if (gid >= ROWS * FDIM / 4) return;
  v4f f = *(const v4f*)(src + (size_t)gid * 4);
  bf16_t* d = dst + (size_t)gid * 4;
  #pragma unroll
  for (int i = 0; i < 4; ++i) d[i] = (bf16_t)f[i];
}

// ---------------- kernel 1: x_proj = x @ w_in^T  (M=32000,K=256,N=512) ----------------
// Each wave: one M-tile x 8 N-tiles -> one A-fragment feeds 8 WMMAs.
__global__ void __launch_bounds__(128) xproj_kernel(const bf16_t* __restrict__ xb,
                                                    const bf16_t* __restrict__ win_pk,
                                                    float* __restrict__ xproj) {
  int wave = threadIdx.x >> 5, lane = threadIdx.x & 31;
  int wid = blockIdx.x * 4 + wave;          // 2000 M-tiles * 4 N-groups = 8000 jobs
  int mt = wid >> 2, ng = wid & 3;          // N-group: 8 tiles
  int mbase = mt * 16;
  v8f c[8];
  #pragma unroll
  for (int q = 0; q < 8; ++q)
    #pragma unroll
    for (int e = 0; e < 8; ++e) c[q][e] = 0.f;

  #pragma unroll
  for (int k0 = 0; k0 < FDIM; k0 += 32) {
    v16bf a = load_a_bf16(xb, FDIM, mbase, k0, lane);
    #pragma unroll
    for (int q = 0; q < 8; ++q) {
      v16bf b = load_b_pk(win_pk, HDIM / 16, k0 >> 5, ng * 8 + q, lane);
      c[q] = wmma_bf16(a, b, c[q]);
    }
  }
  int mrow = mbase + ((lane >> 4) << 3);
  #pragma unroll
  for (int q = 0; q < 8; ++q) {
    int n = (ng * 8 + q) * 16 + (lane & 15);
    #pragma unroll
    for (int j = 0; j < 8; ++j) xproj[(size_t)(mrow + j) * HDIM + n] = c[q][j];
  }
}

// ---------------- kernel 2: persistent LIF-AdEx scan ----------------
// 16 blocks x 128 threads. Block j owns H columns [32j, 32j+32).
// w_rec slice lives in LDS for all 500 steps; v/i/a state lives in LDS.
__global__ void __launch_bounds__(128, 1)
scan_kernel(const float* __restrict__ xproj, const bf16_t* __restrict__ wrec_pk,
            bf16_t* __restrict__ spikes, float* __restrict__ state_out,
            unsigned* __restrict__ sync) {
  __shared__ bf16_t wB[2 * 16 * 512];            // 32 KB: frag (k,f) at ((k*2+f)*512)
  __shared__ float v_s[2048], i_s[2048], a_s[2048];

  const int j = blockIdx.x;                       // 0..15 (n-slice)
  const int tid = threadIdx.x, wave = tid >> 5, lane = tid & 31;
  const int NTrec = HDIM / 16;                    // 32

  // stage this block's two B-tile columns of packed w_rec into LDS (2048 uint4)
  for (int u = tid; u < 2048; u += 128) {
    int q = u >> 6, r = u & 63;                   // frag q (k*2+f), uint4 r within frag
    int k = q >> 1, f = q & 1;
    ((uint4*)wB)[u] =
        ((const uint4*)(wrec_pk + ((size_t)(k * NTrec + 2 * j + f) * 512)))[r];
  }
  for (int u = tid; u < 2048; u += 128) { v_s[u] = 0.f; i_s[u] = 0.f; a_s[u] = 0.f; }
  __syncthreads();

  const int m = wave;                             // M tile (16 rows of B=64)
  unsigned* cnt = sync;
  unsigned* gen = sync + 1;

  for (int t = 0; t < T_STEPS; ++t) {
    v8f c0, c1;
    #pragma unroll
    for (int q = 0; q < 8; ++q) { c0[q] = 0.f; c1[q] = 0.f; }

    if (t > 0) {                                  // R = z_{t-1} @ w_rec^T (slice)
      const bf16_t* zprev = spikes + (size_t)(t - 1) * BATCH * HDIM;
      #pragma unroll 8
      for (int k0 = 0; k0 < HDIM; k0 += 32) {
        v16bf a  = load_a_bf16(zprev, HDIM, m * 16, k0, lane);
        v16bf b0 = *(const v16bf*)&wB[(((k0 >> 5) * 2 + 0) * 512) + lane * 16];
        v16bf b1 = *(const v16bf*)&wB[(((k0 >> 5) * 2 + 1) * 512) + lane * 16];
        c0 = wmma_bf16(a, b0, c0);
        c1 = wmma_bf16(a, b1, c1);
      }
    }

    const float* xp_t = xproj + (size_t)t * BATCH * HDIM;
    bf16_t* z_t = spikes + (size_t)t * BATCH * HDIM;

    #pragma unroll
    for (int f = 0; f < 2; ++f) {
      v8f c = f ? c1 : c0;
      int hl = f * 16 + (lane & 15);              // local column 0..31
      int h  = j * 32 + hl;
      #pragma unroll
      for (int jj = 0; jj < 8; ++jj) {
        int b   = m * 16 + jj + ((lane >> 4) << 3);
        int idx = b * 32 + hl;
        float vv = v_s[idx], ii = i_s[idx], aa = a_s[idx];
        float dv = DTs * TAU_MEM_INV *
                   ((V_LEAK - vv) + DELTA_T * __expf((vv - V_TH) * (1.0f / DELTA_T)) +
                    ii - aa);
        float v_dec = vv + dv;
        float i_dec = ii - DTs * TAU_SYN_INV * ii;
        float a_dec = aa + DTs * TAU_ADA_INV * (ADAPT_CURRENT * (vv - V_LEAK) - aa);
        float z     = (v_dec - V_TH) > 0.f ? 1.f : 0.f;
        float v_new = (1.f - z) * v_dec + z * V_RESET;
        float i_new = i_dec + xp_t[(size_t)b * HDIM + h] + c[jj];
        float a_new = a_dec + z * ADAPT_SPIKE;
        v_s[idx] = v_new; i_s[idx] = i_new; a_s[idx] = a_new;
        z_t[(size_t)b * HDIM + h] = (bf16_t)z;
        if (t == T_STEPS - 1) {
          size_t e = (size_t)b * HDIM + h;
          state_out[0 * (size_t)BATCH * HDIM + e] = z;
          state_out[1 * (size_t)BATCH * HDIM + e] = v_new;
          state_out[2 * (size_t)BATCH * HDIM + e] = i_new;
          state_out[3 * (size_t)BATCH * HDIM + e] = a_new;
        }
      }
    }

    // device-wide barrier: spikes[t] must be globally visible before step t+1
    __threadfence();
    __syncthreads();
    if (tid == 0) {
      unsigned g = __hip_atomic_load(gen, __ATOMIC_RELAXED, __HIP_MEMORY_SCOPE_AGENT);
      unsigned a = __hip_atomic_fetch_add(cnt, 1u, __ATOMIC_ACQ_REL,
                                          __HIP_MEMORY_SCOPE_AGENT);
      if (a == gridDim.x - 1) {
        __hip_atomic_store(cnt, 0u, __ATOMIC_RELAXED, __HIP_MEMORY_SCOPE_AGENT);
        __hip_atomic_fetch_add(gen, 1u, __ATOMIC_RELEASE, __HIP_MEMORY_SCOPE_AGENT);
      } else {
        while (__hip_atomic_load(gen, __ATOMIC_ACQUIRE, __HIP_MEMORY_SCOPE_AGENT) == g)
          __builtin_amdgcn_s_sleep(1);
      }
    }
    __syncthreads();
  }
}

// ---------------- kernel 3: lin = spikes @ w_out^T + b  (M=32000,K=512,N=128) --------
// Each wave: one M-tile x all 8 N-tiles -> spikes read exactly once.
__global__ void __launch_bounds__(128) lin_kernel(const bf16_t* __restrict__ spikes,
                                                  const bf16_t* __restrict__ wout_pk,
                                                  const float* __restrict__ b_out,
                                                  float* __restrict__ lin) {
  int wave = threadIdx.x >> 5, lane = threadIdx.x & 31;
  int mt = blockIdx.x * 4 + wave;              // 2000 M-tiles
  int mbase = mt * 16;
  v8f c[8];
  #pragma unroll
  for (int q = 0; q < 8; ++q)
    #pragma unroll
    for (int e = 0; e < 8; ++e) c[q][e] = 0.f;

  #pragma unroll 4
  for (int k0 = 0; k0 < HDIM; k0 += 32) {
    v16bf a = load_a_bf16(spikes, HDIM, mbase, k0, lane);
    #pragma unroll
    for (int q = 0; q < 8; ++q) {
      v16bf b = load_b_pk(wout_pk, ODIM / 16, k0 >> 5, q, lane);
      c[q] = wmma_bf16(a, b, c[q]);
    }
  }
  int mrow = mbase + ((lane >> 4) << 3);
  #pragma unroll
  for (int q = 0; q < 8; ++q) {
    int n = q * 16 + (lane & 15);
    float bias = b_out[n];
    #pragma unroll
    for (int j = 0; j < 8; ++j) lin[(size_t)(mrow + j) * ODIM + n] = c[q][j] + bias;
  }
}

// ---------------- kernel 4: exponential low-pass filter over T ----------------
__global__ void filter_kernel(const float* __restrict__ lin, float* __restrict__ out) {
  int id = blockIdx.x * 256 + threadIdx.x;     // b*ODIM + o, 8192 total
  if (id >= BATCH * ODIM) return;
  float y = 0.f;
  for (int t = 0; t < T_STEPS; ++t) {
    float l = lin[(size_t)t * BATCH * ODIM + id];
    y = y + DTs * TAU_FILT_INV * (l - y);
    out[(size_t)t * BATCH * ODIM + id] = y;
  }
}

__global__ void init_sync_kernel(unsigned* sync) { sync[0] = 0u; }

// ---------------- host launcher ----------------
extern "C" void kernel_launch(void* const* d_in, const int* in_sizes, int n_in,
                              void* d_out, int out_size, void* d_ws, size_t ws_size,
                              hipStream_t stream) {
  const float* x     = (const float*)d_in[0];   // [500,64,256]
  const float* w_in  = (const float*)d_in[1];   // [512,256]
  const float* w_rec = (const float*)d_in[2];   // [512,512]
  const float* w_out = (const float*)d_in[3];   // [128,512]
  const float* b_out = (const float*)d_in[4];   // [128]

  char* ws = (char*)d_ws;
  bf16_t*   win_pk  = (bf16_t*)(ws + OFF_WINP);
  bf16_t*   wrec_pk = (bf16_t*)(ws + OFF_WRECP);
  bf16_t*   wout_pk = (bf16_t*)(ws + OFF_WOUTP);
  bf16_t*   xb      = (bf16_t*)(ws + OFF_XB);
  float*    xproj   = (float*)(ws + OFF_XPROJ);
  bf16_t*   spikes  = (bf16_t*)(ws + OFF_SPK);
  float*    lin     = (float*)(ws + OFF_LIN);
  unsigned* sync    = (unsigned*)(ws + OFF_SYNC);

  float* out       = (float*)d_out;                           // [500,64,128]
  float* state_out = out + (size_t)T_STEPS * BATCH * ODIM;    // z,v,i,a [64,512] each

  init_sync_kernel<<<1, 1, 0, stream>>>(sync);

  pack_b_kernel<<<(FDIM * HDIM + 255) / 256, 256, 0, stream>>>(w_in, win_pk, FDIM, HDIM);
  pack_b_kernel<<<(HDIM * HDIM + 255) / 256, 256, 0, stream>>>(w_rec, wrec_pk, HDIM, HDIM);
  pack_b_kernel<<<(HDIM * ODIM + 255) / 256, 256, 0, stream>>>(w_out, wout_pk, HDIM, ODIM);
  cvt_x_kernel<<<(ROWS * FDIM / 4 + 255) / 256, 256, 0, stream>>>(x, xb);

  // 2000 M-tiles x 4 N-groups (8 tiles each), 4 wave-jobs per block
  xproj_kernel<<<2000, 128, 0, stream>>>(xb, win_pk, xproj);

  scan_kernel<<<16, 128, 0, stream>>>(xproj, wrec_pk, spikes, state_out, sync);

  // 2000 M-tiles, each wave does all 8 N-tiles
  lin_kernel<<<500, 128, 0, stream>>>(spikes, wout_pk, b_out, lin);

  filter_kernel<<<(BATCH * ODIM + 255) / 256, 256, 0, stream>>>(lin, out);
}
